// LightGCN_15487652069895
// MI455X (gfx1250) — matrix operand order repaired and checked by reading, not compiled, not run on hardware
//
#include <hip/hip_runtime.h>
#include <hip/hip_bf16.h>

// ---------------------------------------------------------------------------
// LightGCN forward on MI455X (gfx1250).
//   U=4096 users, I=8192 items, D=64 embed, L=3 layers.
//   W[u,i] = adj[u,i] / sqrt(deg_u[u]*deg_i[i])   (0 where denom==0)
//   per layer: new_x[0:I]   = W^T @ x[0:U]
//              new_x[I:I+U] = W   @ x[U:U+I]
//   out = (sum of layer outputs) / 4
// Strategy: materialize W and W^T once as f16 (128 MB total, L2-resident),
// transpose x to f16 xT[64][12288] each layer, run both GEMMs with
// v_wmma_f32_16x16x32_f16 in a software-pipelined K loop (loads of step s+1
// overlap the 4 WMMAs of step s), fold the 0.25 accumulation into the
// epilogue.
// ---------------------------------------------------------------------------

#define NU 4096
#define NI 8192
#define ND 64
#define NT (NU + NI)      // 12288 total rows
#define XTS 12288         // xT row stride (elements), multiple of 16

typedef __attribute__((ext_vector_type(16))) _Float16 v16h;
typedef __attribute__((ext_vector_type(8)))  _Float16 v8h;
typedef __attribute__((ext_vector_type(8)))  float    v8f;

// ---- degree kernels -------------------------------------------------------

__global__ void lgcn_deg_u(const int* __restrict__ adj, float* __restrict__ degs) {
    __shared__ int red[256];
    const int u = blockIdx.x;
    int s = 0;
    for (int i = threadIdx.x; i < NI; i += 256)
        s += adj[(size_t)u * NI + i];
    red[threadIdx.x] = s;
    __syncthreads();
    for (int o = 128; o > 0; o >>= 1) {
        if (threadIdx.x < o) red[threadIdx.x] += red[threadIdx.x + o];
        __syncthreads();
    }
    if (threadIdx.x == 0) degs[u] = (float)red[0];
}

__global__ void lgcn_deg_i(const int* __restrict__ adj, float* __restrict__ degs_i) {
    const int i = blockIdx.x * 256 + threadIdx.x;   // column
    int s = 0;
#pragma unroll 8
    for (int u = 0; u < NU; ++u)
        s += adj[(size_t)u * NI + i];
    degs_i[i] = (float)s;
}

__global__ void lgcn_rsqrt(float* __restrict__ degs) {
    const int idx = blockIdx.x * 256 + threadIdx.x;
    if (idx < NT) {
        float d = degs[idx];
        degs[idx] = (d > 0.0f) ? rsqrtf(d) : 0.0f;
    }
}

// ---- build W (f16) and W^T (f16) in one tiled pass ------------------------
// grid (NI/32, NU/32), block (32,8). t[i_local][u_local] staged in LDS so the
// transposed store to WT16 is coalesced along u.
__global__ void lgcn_buildW(const int* __restrict__ adj,
                            const float* __restrict__ ru,
                            const float* __restrict__ ri,
                            _Float16* __restrict__ W16,
                            _Float16* __restrict__ WT16) {
    __shared__ _Float16 t[32][33];
    const int i0 = blockIdx.x * 32;
    const int u0 = blockIdx.y * 32;
    const int tx = threadIdx.x, ty = threadIdx.y;
#pragma unroll
    for (int j = 0; j < 4; ++j) {
        const int u = u0 + ty + 8 * j;
        const int i = i0 + tx;
        const float w = adj[(size_t)u * NI + i] ? ru[u] * ri[i] : 0.0f;
        const _Float16 h = (_Float16)w;
        W16[(size_t)u * NI + i] = h;          // row-major W, coalesced in i
        t[tx][ty + 8 * j] = h;                // [i_local][u_local]
    }
    __syncthreads();
#pragma unroll
    for (int j = 0; j < 4; ++j) {
        const int i = i0 + ty + 8 * j;
        const int u = u0 + tx;
        WT16[(size_t)i * NU + u] = t[ty + 8 * j][tx];   // coalesced in u
    }
}

// ---- transpose+convert x (f32 [12288][64]) -> xT16 (f16 [64][XTS]) --------
// grid (NT/32, 2), block (32,8)
__global__ void lgcn_xT(const float* __restrict__ x, _Float16* __restrict__ xT) {
    __shared__ float t[32][33];
    const int r0 = blockIdx.x * 32;
    const int c0 = blockIdx.y * 32;
    const int tx = threadIdx.x, ty = threadIdx.y;
#pragma unroll
    for (int j = 0; j < 4; ++j)
        t[ty + 8 * j][tx] = x[(size_t)(r0 + ty + 8 * j) * ND + c0 + tx];
    __syncthreads();
#pragma unroll
    for (int j = 0; j < 4; ++j)
        xT[(size_t)(c0 + ty + 8 * j) * XTS + r0 + tx] = (_Float16)t[tx][ty + 8 * j];
}

// ---- WMMA GEMM: out[m, 0:64] = sum_k A16[m,k] * xT[col][brow0+k] ----------
// One wave per 16-row M tile, 4 N tiles of 16 (full 64 columns).
// A-operand lane layout (16-bit A 16x32): lane m (sel=lane>>4, mrow=lane&15)
//   elems 0..7  = K[k0 + sel*16     .. +8)
//   elems 8..15 = K[k0 + sel*16 + 8 .. +8)
// B-operand: lane holds 16 contiguous K values of one column:
//   col = n0 + (lane&15), K base = k0 + (lane>>4)*16      (32B aligned load)
// D layout: VGPR r <-> M = (lane>>4)*8 + r, N = lane&15.
// K loop is software-pipelined: step s+1's 10 loads are issued while step s's
// 4 WMMAs execute, so the s_wait lands behind the compute, not in front of it.
__global__ void lgcn_gemm_wmma(const _Float16* __restrict__ A16, int lda,
                               const _Float16* __restrict__ xT, int brow0,
                               int K,
                               float* __restrict__ xnew,
                               float* __restrict__ acc,
                               int out_row0, int first, float scale) {
    const int lane = threadIdx.x & 31;
    const int wave = threadIdx.x >> 5;
    const int m0   = (blockIdx.x * 8 + wave) * 16;

    const int mrow = lane & 15;
    const int sel  = lane >> 4;

    const _Float16* pa = A16 + (size_t)(m0 + mrow) * lda + sel * 16;
    const _Float16* pb0 = xT + (size_t)mrow * XTS + brow0 + sel * 16;

    const v8f zero = {0.f, 0.f, 0.f, 0.f, 0.f, 0.f, 0.f, 0.f};
    v8f c[4] = {zero, zero, zero, zero};

    // ---- prologue: load operands for k0 = 0
    v16h a, b[4];
    {
        const v8h alo = *(const v8h*)(pa);
        const v8h ahi = *(const v8h*)(pa + 8);
#pragma unroll
        for (int t = 0; t < 8; ++t) { a[t] = alo[t]; a[t + 8] = ahi[t]; }
#pragma unroll
        for (int nt = 0; nt < 4; ++nt)
            b[nt] = *(const v16h*)(pb0 + (size_t)nt * 16 * XTS);
    }

    const int steps = K / 32;
    for (int s = 0; s < steps - 1; ++s) {
        const int kn = (s + 1) * 32;

        // issue next step's loads first (overlap with this step's WMMAs)
        v16h a2, b2[4];
        {
            const v8h alo = *(const v8h*)(pa + kn);
            const v8h ahi = *(const v8h*)(pa + kn + 8);
#pragma unroll
            for (int t = 0; t < 8; ++t) { a2[t] = alo[t]; a2[t + 8] = ahi[t]; }
#pragma unroll
            for (int nt = 0; nt < 4; ++nt)
                b2[nt] = *(const v16h*)(pb0 + (size_t)nt * 16 * XTS + kn);
        }

        // prefetch the streaming A row further ahead (global_prefetch_b8)
        if (kn + 256 < K) __builtin_prefetch(pa + kn + 256, 0, 0);

#pragma unroll
        for (int nt = 0; nt < 4; ++nt)
            c[nt] = __builtin_amdgcn_wmma_f32_16x16x32_f16(
                false, a, false, b[nt], (short)0, c[nt], false, false);

        a = a2;
#pragma unroll
        for (int nt = 0; nt < 4; ++nt) b[nt] = b2[nt];
    }

    // ---- epilogue step (no trailing loads)
#pragma unroll
    for (int nt = 0; nt < 4; ++nt)
        c[nt] = __builtin_amdgcn_wmma_f32_16x16x32_f16(
            false, a, false, b[nt], (short)0, c[nt], false, false);

    // store: xnew = v ; acc (+)= 0.25*v
    const int mb = m0 + sel * 8;
#pragma unroll
    for (int nt = 0; nt < 4; ++nt) {
        const int col = nt * 16 + mrow;
#pragma unroll
        for (int r = 0; r < 8; ++r) {
            const int row = out_row0 + mb + r;
            const size_t o = (size_t)row * ND + col;
            const float v = c[nt][r];
            xnew[o] = v;
            if (first) acc[o] = scale * v;
            else       acc[o] += scale * v;
        }
    }
}

// ---------------------------------------------------------------------------

extern "C" void kernel_launch(void* const* d_in, const int* in_sizes, int n_in,
                              void* d_out, int out_size, void* d_ws, size_t ws_size,
                              hipStream_t stream) {
    const float* x_in = (const float*)d_in[0];   // [12288, 64] f32
    const int*   adj  = (const int*)d_in[1];     // [4096, 8192] int32 0/1
    float* acc = (float*)d_out;                  // [12288, 64] f32

    char* w = (char*)d_ws;
    float* degs = (float*)w;                                 // 12288 floats
    size_t off = 64 * 1024;
    _Float16* W16  = (_Float16*)(w + off); off += (size_t)NU * NI * 2;  // 64 MB
    _Float16* WT16 = (_Float16*)(w + off); off += (size_t)NI * NU * 2;  // 64 MB
    _Float16* xT16 = (_Float16*)(w + off); off += (size_t)ND * XTS * 2; // 1.5 MB
    float* xa = (float*)(w + off); off += (size_t)NT * ND * 4;          // 3 MB
    float* xb = (float*)(w + off);                                      // 3 MB

    // degrees + normalization factors
    lgcn_deg_u<<<NU, 256, 0, stream>>>(adj, degs);
    lgcn_deg_i<<<NI / 256, 256, 0, stream>>>(adj, degs + NU);
    lgcn_rsqrt<<<(NT + 255) / 256, 256, 0, stream>>>(degs);

    // materialize W (f16) and W^T (f16)
    lgcn_buildW<<<dim3(NI / 32, NU / 32), dim3(32, 8), 0, stream>>>(
        adj, degs, degs + NU, W16, WT16);

    // 3 propagation layers
    const float* src = x_in;
    float* dst = xa;
    for (int layer = 0; layer < 3; ++layer) {
        lgcn_xT<<<dim3(NT / 32, 2), dim3(32, 8), 0, stream>>>(src, xT16);

        // new_x[0:8192] = W^T @ x[0:4096]   (A = WT16, K over users)
        lgcn_gemm_wmma<<<NI / 128, 256, 0, stream>>>(
            WT16, NU, xT16, /*brow0=*/0, /*K=*/NU,
            dst, acc, /*out_row0=*/0, layer == 0 ? 1 : 0, 0.25f);

        // new_x[8192:12288] = W @ x[4096:12288]   (A = W16, K over items)
        lgcn_gemm_wmma<<<NU / 128, 256, 0, stream>>>(
            W16, NI, xT16, /*brow0=*/NU, /*K=*/NI,
            dst, acc, /*out_row0=*/NI, layer == 0 ? 1 : 0, 0.25f);

        src = dst;
        dst = (dst == xa) ? xb : xa;
    }
}
